// BidirectionalTemporalEncoder_28441273434660
// MI455X (gfx1250) — matrix operand order repaired and checked by reading, not compiled, not run on hardware
//
#include <hip/hip_runtime.h>
#include <hip/hip_bf16.h>

// ---------------------------------------------------------------------------
// MI455X (gfx1250) implementation.
//  - All GEMMs run on the matrix pipe via v_wmma_f32_16x16x32_bf16
//    (fp32 -> bf16 in-register, fp32 accumulate). fp32 WMMA (16x16x4) has 8x
//    lower K-throughput, so bf16 is the chip-rewarded path.
//  - GEMM inner loop is branch-free: addresses are clamped (OOB A rows / B
//    cols only feed C elements that are never stored) and the K-tail uses
//    data selects, so no EXEC-mask diamonds / serialized loads.
//  - A fragments (and transposed-B fragments) are per-lane contiguous in the
//    k order used here (k = 16*half + e, not AMD's table order) -> global_load_b128 vectorized.
//  - Each wave computes a 64x16 C panel (4 WMMAs share one B fragment).
// ---------------------------------------------------------------------------

typedef __attribute__((ext_vector_type(16))) __bf16 v16bf;
typedef __attribute__((ext_vector_type(8)))  float  v8f;

namespace {
constexpr int CB = 8;      // batch
constexpr int CT = 400;    // seq len
constexpr int CE = 512;    // emb == hid
constexpr int CH = 8;      // heads
constexpr int CD = 64;     // head dim
constexpr int CL = 399;    // transformer seq len (T-1)
constexpr int CF = 1024;   // FF dim
constexpr int SCP = 400;   // score row pitch (16B-aligned rows for b128 loads)
constexpr int NBT = CB * CT;   // 3200 rows
constexpr int NBL = CB * CL;   // 3192 rows
}

// ---------------------------------------------------------------------------
// Batched GEMM: C[z] = act( A[z](MxK) * B[z](KxN or NxK^T) + bias )
// One wave per 64x16 C panel; K stepped by 32.
// ---------------------------------------------------------------------------
__global__ __launch_bounds__(128) void k_gemm_wmma_bf16(
    const float* __restrict__ A, const float* __restrict__ Bm,
    const float* __restrict__ bias, float* __restrict__ C,
    int M, int N, int K, int lda, int ldb, int ldc,
    int transB, int act, int nb2,
    long long sA1, long long sA2, long long sB1, long long sB2,
    long long sC1, long long sC2)
{
    const int z  = blockIdx.z;
    const int z1 = z / nb2, z2 = z - z1 * nb2;
    A  += z1 * sA1 + z2 * sA2;
    Bm += z1 * sB1 + z2 * sB2;
    C  += z1 * sC1 + z2 * sC2;

    const int tilesN  = (N + 15) >> 4;
    const int tilesM4 = (M + 63) >> 6;
    const int group   = blockIdx.x * 4 + ((int)threadIdx.x >> 5);
    if (group >= tilesM4 * tilesN) return;   // wave-uniform exit

    const int lane = (int)threadIdx.x & 31;
    const int half = lane >> 4;
    const int l16  = lane & 15;
    const int row0 = (group / tilesN) << 6;
    const int col0 = (group % tilesN) << 4;
    const int bn   = col0 + l16;
    const int bnc  = (bn < N) ? bn : (N - 1);   // clamped: OOB col never stored

    const float* ap[4];
#pragma unroll
    for (int i = 0; i < 4; ++i) {
        int am = row0 + i * 16 + l16;
        if (am >= M) am = M - 1;                // clamped: OOB row never stored
        ap[i] = A + (size_t)am * lda;
    }
    const float* bp = transB ? (Bm + (size_t)bnc * ldb) : (Bm + bnc);

    v8f acc[4] = {v8f{}, v8f{}, v8f{}, v8f{}};

    const int Kfull = K & ~31;
    for (int k0 = 0; k0 < Kfull; k0 += 32) {
        // ---- B fragment: element e <-> k = k0 + 16*half + e ----
        v16bf b;
        if (transB) {   // per-lane contiguous: 4x b128
            const float4* q = (const float4*)(bp + k0 + (half << 4));
            const float4 b0 = q[0], b1 = q[1], b2 = q[2], b3 = q[3];
            b[0]=(__bf16)b0.x;  b[1]=(__bf16)b0.y;  b[2]=(__bf16)b0.z;  b[3]=(__bf16)b0.w;
            b[4]=(__bf16)b1.x;  b[5]=(__bf16)b1.y;  b[6]=(__bf16)b1.z;  b[7]=(__bf16)b1.w;
            b[8]=(__bf16)b2.x;  b[9]=(__bf16)b2.y;  b[10]=(__bf16)b2.z; b[11]=(__bf16)b2.w;
            b[12]=(__bf16)b3.x; b[13]=(__bf16)b3.y; b[14]=(__bf16)b3.z; b[15]=(__bf16)b3.w;
        } else {        // wave-coalesced column walk
#pragma unroll
            for (int e = 0; e < 16; ++e)
                b[e] = (__bf16)bp[(size_t)(k0 + (half << 4) + e) * ldb];
        }
        // ---- 4 A fragments, one WMMA each (B reused) ----
#pragma unroll
        for (int i = 0; i < 4; ++i) {
            // A fragment = one contiguous 16-float run: k0+16h..
            const float4* q0 = (const float4*)(ap[i] + k0 + (half << 4));
            const float4* q1 = (const float4*)(ap[i] + k0 + (half << 4) + 8);
            const float4 a0 = q0[0], a1 = q0[1], a2 = q1[0], a3 = q1[1];
            v16bf a;
            a[0]=(__bf16)a0.x;  a[1]=(__bf16)a0.y;  a[2]=(__bf16)a0.z;  a[3]=(__bf16)a0.w;
            a[4]=(__bf16)a1.x;  a[5]=(__bf16)a1.y;  a[6]=(__bf16)a1.z;  a[7]=(__bf16)a1.w;
            a[8]=(__bf16)a2.x;  a[9]=(__bf16)a2.y;  a[10]=(__bf16)a2.z; a[11]=(__bf16)a2.w;
            a[12]=(__bf16)a3.x; a[13]=(__bf16)a3.y; a[14]=(__bf16)a3.z; a[15]=(__bf16)a3.w;
            acc[i] = __builtin_amdgcn_wmma_f32_16x16x32_bf16(
                         false, a, false, b, (short)0, acc[i], false, false);
        }
    }

    // ---- K tail (only GEMMs with K%32 != 0, e.g. P@V with K=399) ----
    if (Kfull < K) {
        const int k0 = Kfull;
        v16bf b;
#pragma unroll
        for (int e = 0; e < 16; ++e) {
            const int kb = k0 + (half << 4) + e;
            const int kc = (kb < K) ? kb : (K - 1);
            const float bv = transB ? bp[kc] : bp[(size_t)kc * ldb];
            b[e] = (kb < K) ? (__bf16)bv : (__bf16)0.f;   // data select, no branch
        }
#pragma unroll
        for (int i = 0; i < 4; ++i) {
            v16bf a;
#pragma unroll
            for (int e = 0; e < 16; ++e) {
                const int ka = k0 + (half << 4) + e;
                const int kc = (ka < K) ? ka : (K - 1);
                const float av = ap[i][kc];
                a[e] = (ka < K) ? (__bf16)av : (__bf16)0.f;
            }
            acc[i] = __builtin_amdgcn_wmma_f32_16x16x32_bf16(
                         false, a, false, b, (short)0, acc[i], false, false);
        }
    }

    // ---- epilogue: bias + activation + guarded store ----
#pragma unroll
    for (int i = 0; i < 4; ++i) {
#pragma unroll
        for (int v = 0; v < 8; ++v) {
            const int mr = row0 + i * 16 + (half << 3) + v;
            if (mr < M && bn < N) {
                float r = acc[i][v];
                if (bias) r += bias[bn];
                if (act == 1)      r = fmaxf(r, 0.f);
                else if (act == 2) r = 0.5f * r * (1.f + erff(r * 0.70710678118f));
                C[(size_t)mr * ldc + bn] = r;
            }
        }
    }
}

// ---- BatchNorm over rows (biased var, matches jnp.var) --------------------
__global__ __launch_bounds__(256) void k_bnstats(
    const float* __restrict__ X, int M, int N,
    float* __restrict__ mean, float* __restrict__ var)
{
    const int j = blockIdx.x * blockDim.x + threadIdx.x;
    if (j >= N) return;
    float s = 0.f, s2 = 0.f;
    for (int i = 0; i < M; ++i) {
        const float v = X[(size_t)i * N + j];
        s += v; s2 += v * v;
    }
    const float mu = s / (float)M;
    mean[j] = mu;
    var[j]  = s2 / (float)M - mu * mu;
}

__global__ __launch_bounds__(256) void k_bnapply(
    const float* __restrict__ X, const float* __restrict__ g,
    const float* __restrict__ be, const float* __restrict__ mean,
    const float* __restrict__ var, float* __restrict__ Y,
    int total, int N, int relu)
{
    const int i = blockIdx.x * blockDim.x + threadIdx.x;
    if (i >= total) return;
    const int j = i % N;
    float r = g[j] * (X[i] - mean[j]) * rsqrtf(var[j] + 1e-5f) + be[j];
    if (relu) r = fmaxf(r, 0.f);
    Y[i] = r;
}

// ---- LayerNorm over last dim (512) with optional fused residual -----------
__global__ __launch_bounds__(256) void k_ln512(
    const float* __restrict__ X, const float* __restrict__ R,
    const float* __restrict__ g, const float* __restrict__ be,
    float* __restrict__ Y, int rows)
{
    const int row = (blockIdx.x * blockDim.x + threadIdx.x) >> 5;
    if (row >= rows) return;
    const int lane = (int)threadIdx.x & 31;
    const float* xr = X + (size_t)row * 512;
    const float* rr = R ? (R + (size_t)row * 512) : nullptr;
    float v[16];
    float s = 0.f, s2 = 0.f;
#pragma unroll
    for (int i = 0; i < 16; ++i) {
        const int j = lane + i * 32;
        float t = xr[j];
        if (rr) t += rr[j];
        v[i] = t; s += t; s2 += t * t;
    }
#pragma unroll
    for (int o = 16; o; o >>= 1) { s += __shfl_xor(s, o); s2 += __shfl_xor(s2, o); }
    const float mu  = s  * (1.f / 512.f);
    const float var = s2 * (1.f / 512.f) - mu * mu;
    const float rs  = rsqrtf(var + 1e-5f);
    float* yr = Y + (size_t)row * 512;
#pragma unroll
    for (int i = 0; i < 16; ++i) {
        const int j = lane + i * 32;
        yr[j] = g[j] * (v[i] - mu) * rs + be[j];
    }
}

// ---- Causal softmax over padded rows; mask with -1e20 before /sqrt(64) ----
__global__ __launch_bounds__(256) void k_softmax_causal(
    float* __restrict__ S, int rows, int L, int ldp)
{
    const int r = (blockIdx.x * blockDim.x + threadIdx.x) >> 5;
    if (r >= rows) return;
    const int lane = (int)threadIdx.x & 31;
    const int q = r % L;
    float* p = S + (size_t)r * ldp;
    float m = -3.4e38f;
    for (int k = lane; k <= q; k += 32) m = fmaxf(m, p[k] * 0.125f);
#pragma unroll
    for (int o = 16; o; o >>= 1) m = fmaxf(m, __shfl_xor(m, o));
    float s = 0.f;
    for (int k = lane; k < L; k += 32) {
        float e = 0.f;
        if (k <= q) e = expf(p[k] * 0.125f - m);
        p[k] = e; s += e;
    }
#pragma unroll
    for (int o = 16; o; o >>= 1) s += __shfl_xor(s, o);
    const float inv = 1.f / s;
    for (int k = lane; k <= q; k += 32) p[k] *= inv;
}

// ---- elementwise / layout kernels -----------------------------------------
__global__ __launch_bounds__(256) void k_addpos(
    const float* __restrict__ X, const float* __restrict__ POS,
    float* __restrict__ Y, int total)
{
    const int i = blockIdx.x * blockDim.x + threadIdx.x;
    if (i >= total) return;
    Y[i] = X[i] + POS[i % (CL * CE)];
}

__global__ __launch_bounds__(256) void k_slice_f(
    const float* __restrict__ P, float* __restrict__ O, int total)
{
    const int i = blockIdx.x * blockDim.x + threadIdx.x;
    if (i >= total) return;
    const int b = i / (CL * CE), r = i % (CL * CE);
    O[i] = P[(size_t)b * CT * CE + r];                 // proj[:, :-1, :]
}

__global__ __launch_bounds__(256) void k_slice_b(
    const float* __restrict__ P, float* __restrict__ O, int total)
{
    const int i = blockIdx.x * blockDim.x + threadIdx.x;
    if (i >= total) return;
    const int b = i / (CL * CE), r = i % (CL * CE);
    const int t = r / CE, d = r % CE;
    O[i] = P[((size_t)b * CT + (CL - t)) * CE + d];    // flip(proj[:,1:,:])
}

__global__ __launch_bounds__(256) void k_assemble_f(
    const float* __restrict__ X, const float* __restrict__ FY,
    float* __restrict__ O, int total)
{
    const int i = blockIdx.x * blockDim.x + threadIdx.x;
    if (i >= total) return;
    const int b = i / (CT * CE), r = i % (CT * CE);
    const int t = r / CE, d = r % CE;
    O[i] = (t == 0) ? X[(size_t)b * CT * CE + d]
                    : FY[((size_t)b * CL + (t - 1)) * CE + d];
}

__global__ __launch_bounds__(256) void k_assemble_b(
    const float* __restrict__ X, const float* __restrict__ BY,
    float* __restrict__ O, int total)
{
    const int i = blockIdx.x * blockDim.x + threadIdx.x;
    if (i >= total) return;
    const int b = i / (CT * CE), r = i % (CT * CE);
    const int t = r / CE, d = r % CE;
    O[i] = (t == CT - 1) ? X[((size_t)b * CT + (CT - 1)) * CE + d]
                         : BY[((size_t)b * CL + (CL - 1 - t)) * CE + d];
}

__global__ __launch_bounds__(256) void k_avg(
    const float* __restrict__ A, const float* __restrict__ Bp,
    float* __restrict__ O, int total)
{
    const int i = blockIdx.x * blockDim.x + threadIdx.x;
    if (i >= total) return;
    O[i] = 0.5f * (A[i] + Bp[i]);
}

__global__ __launch_bounds__(256) void k_rowdot(
    const float* __restrict__ X, const float* __restrict__ w,
    const float* __restrict__ b, float* __restrict__ out, int rows, int N)
{
    const int row = (blockIdx.x * blockDim.x + threadIdx.x) >> 5;
    if (row >= rows) return;
    const int lane = (int)threadIdx.x & 31;
    float s = 0.f;
    for (int j = lane; j < N; j += 32) s += X[(size_t)row * N + j] * w[j];
#pragma unroll
    for (int o = 16; o; o >>= 1) s += __shfl_xor(s, o);
    if (lane == 0) out[row] = s + b[0];
}

__global__ __launch_bounds__(256) void k_clipcomb(
    const float* __restrict__ F, const float* __restrict__ w,
    float* __restrict__ O, int total, int N)
{
    const int i = blockIdx.x * blockDim.x + threadIdx.x;
    if (i >= total) return;
    const int g = i / N, d = i % N;
    float s = 0.f;
#pragma unroll
    for (int c = 0; c < 5; ++c) s += F[((size_t)g * 5 + c) * N + d] * w[g * 5 + c];
    O[(size_t)g * N + d] = s;
}

// ---------------------------------------------------------------------------
extern "C" void kernel_launch(void* const* d_in, const int* in_sizes, int n_in,
                              void* d_out, int out_size, void* d_ws, size_t ws_size,
                              hipStream_t stream)
{
    (void)in_sizes; (void)n_in; (void)out_size; (void)ws_size;

    // setup_inputs() insertion-order flattening:
    // 0:x | proj1: 1..8 | f_tr: 9..43 | b_tr: 44..78 | proj2: 79..84
    // fusion: 85..90 | fb_ta: 91..92 | proj3: 93..98
    auto IN = [&](int i) -> const float* { return (const float*)d_in[i]; };
    const float* x = IN(0);

    float* dout     = (float*)d_out;
    float* out_fy   = dout;
    float* out_by   = out_fy + (size_t)NBT * CE;
    float* out_fb   = out_by + (size_t)NBT * CE;
    float* out_feat = out_fb + (size_t)NBT * CE;

    float* ws  = (float*)d_ws;
    size_t off = 0;
    auto alloc = [&](size_t n) { float* p = ws + off; off += (n + 63) & ~(size_t)63; return p; };
    float* bufA = alloc((size_t)NBT * CE);
    float* bufB = alloc((size_t)NBT * CE);
    float* bufC = alloc((size_t)NBT * CE);
    float* FINb = alloc((size_t)NBL * CE);
    float* BINb = alloc((size_t)NBL * CE);
    float* Yb   = alloc((size_t)NBL * CE);
    float* Qb   = alloc((size_t)NBL * CE);
    float* Kb   = alloc((size_t)NBL * CE);
    float* Vb   = alloc((size_t)NBL * CE);
    float* ATT  = alloc((size_t)NBL * CE);
    float* ABF  = alloc((size_t)NBL * CE);
    float* TMP  = alloc((size_t)NBL * CE);
    float* FFB  = alloc((size_t)NBL * CF);
    float* SC   = alloc((size_t)CB * CH * CL * SCP);
    float* MEAN = alloc(512);
    float* VAR  = alloc(512);
    float* WTA  = alloc(NBT);
    float* FEAT = alloc((size_t)NBT * CE);

    auto egrid = [](long long n) { return dim3((unsigned)((n + 255) / 256)); };

    auto gemm = [&](const float* A, const float* Bm, const float* bias, float* C,
                    int M, int N, int K, int lda, int ldb, int ldc,
                    int transB, int act, int nbatch, int nb2,
                    long long sA1, long long sA2, long long sB1, long long sB2,
                    long long sC1, long long sC2) {
        const int groups = ((M + 63) / 64) * ((N + 15) / 16);
        dim3 grid((unsigned)((groups + 3) / 4), 1, (unsigned)nbatch);
        k_gemm_wmma_bf16<<<grid, 128, 0, stream>>>(
            A, Bm, bias, C, M, N, K, lda, ldb, ldc, transB, act, nb2,
            sA1, sA2, sB1, sB2, sC1, sC2);
    };
    auto gemm1 = [&](const float* A, const float* Bm, const float* bias, float* C,
                     int M, int N, int K, int act) {
        gemm(A, Bm, bias, C, M, N, K, K, N, N, 0, act, 1, 1, 0, 0, 0, 0, 0, 0);
    };
    auto ln = [&](const float* X, const float* R, const float* g, const float* be,
                  float* Y, int rows) {
        k_ln512<<<dim3((unsigned)(((long long)rows * 32 + 255) / 256)), 256, 0, stream>>>(
            X, R, g, be, Y, rows);
    };
    // 512->512 BN-ReLU MLP (optionally BN on the output)
    auto mlp2 = [&](const float* inp, int M,
                    const float* W1, const float* b1, const float* g1, const float* be1,
                    const float* W2, const float* b2, const float* g2, const float* be2,
                    float* tmp1, float* tmp2, float* outp) {
        gemm1(inp, W1, b1, tmp1, M, 512, 512, 0);
        k_bnstats<<<dim3(2), 256, 0, stream>>>(tmp1, M, 512, MEAN, VAR);
        k_bnapply<<<egrid((long long)M * 512), 256, 0, stream>>>(
            tmp1, g1, be1, MEAN, VAR, tmp2, M * 512, 512, 1);
        if (g2) {
            gemm1(tmp2, W2, b2, tmp1, M, 512, 512, 0);
            k_bnstats<<<dim3(2), 256, 0, stream>>>(tmp1, M, 512, MEAN, VAR);
            k_bnapply<<<egrid((long long)M * 512), 256, 0, stream>>>(
                tmp1, g2, be2, MEAN, VAR, outp, M * 512, 512, 0);
        } else {
            gemm1(tmp2, W2, b2, outp, M, 512, 512, 0);
        }
    };

    auto transformer = [&](int base, const float* inp, float* outp) {
        k_addpos<<<egrid((long long)NBL * CE), 256, 0, stream>>>(
            inp, IN(base), Yb, NBL * CE);
        int li = base + 1;
        for (int layer = 0; layer < 2; ++layer, li += 16) {
            // per-head q/k/v: (B*L*H, 64) @ (64,64)
            gemm1(Yb, IN(li + 0), IN(li + 1), Qb, NBL * CH, CD, CD, 0);
            gemm1(Yb, IN(li + 2), IN(li + 3), Kb, NBL * CH, CD, CD, 0);
            gemm1(Yb, IN(li + 4), IN(li + 5), Vb, NBL * CH, CD, CD, 0);
            // scores[b,h] = Q_bh (L,64) @ K_bh^T  -> (B,H,L,SCP-padded L)
            gemm(Qb, Kb, nullptr, SC, CL, CL, CD, CE, CE, SCP, 1, 0,
                 CB * CH, CH,
                 (long long)CL * CE, CD, (long long)CL * CE, CD,
                 (long long)CH * CL * SCP, (long long)CL * SCP);
            k_softmax_causal<<<dim3((unsigned)(((long long)CB * CH * CL * 32 + 255) / 256)),
                               256, 0, stream>>>(SC, CB * CH * CL, CL, SCP);
            // context[b,h] = P (L,L) @ V_bh (L,64)  -> (B,L,H,D)
            gemm(SC, Vb, nullptr, ATT, CL, CD, CL, SCP, CE, CE, 0, 0,
                 CB * CH, CH,
                 (long long)CH * CL * SCP, (long long)CL * SCP,
                 (long long)CL * CE, CD, (long long)CL * CE, CD);
            gemm1(ATT, IN(li + 6), IN(li + 7), TMP, NBL, CE, CE, 0);
            ln(TMP, Yb, IN(li + 8), IN(li + 9), ABF, NBL);              // ln1(attn+y)
            gemm1(ABF, IN(li + 12), IN(li + 13), FFB, NBL, CF, CE, 2);  // ff1+GELU
            gemm1(FFB, IN(li + 14), IN(li + 15), TMP, NBL, CE, CF, 0);  // ff2
            ln(TMP, ABF, IN(li + 10), IN(li + 11), Yb, NBL);            // ln2(ff+a)
        }
        ln(Yb, nullptr, IN(base + 33), IN(base + 34), outp, NBL);       // lnf
    };

    // ---- proj1 (with bn2) -> proj in bufC ----
    mlp2(x, NBT, IN(1), IN(2), IN(3), IN(4), IN(5), IN(6), IN(7), IN(8),
         bufA, bufB, bufC);
    k_slice_f<<<egrid((long long)NBL * CE), 256, 0, stream>>>(bufC, FINb, NBL * CE);
    k_slice_b<<<egrid((long long)NBL * CE), 256, 0, stream>>>(bufC, BINb, NBL * CE);

    // ---- forward branch ----
    transformer(9, FINb, FINb);
    mlp2(FINb, NBL, IN(79), IN(80), IN(81), IN(82), IN(83), IN(84),
         nullptr, nullptr, bufA, bufB, TMP);
    k_assemble_f<<<egrid((long long)NBT * CE), 256, 0, stream>>>(x, TMP, out_fy, NBT * CE);

    // ---- backward branch ----
    transformer(44, BINb, BINb);
    mlp2(BINb, NBL, IN(79), IN(80), IN(81), IN(82), IN(83), IN(84),
         nullptr, nullptr, bufA, bufB, TMP);
    k_assemble_b<<<egrid((long long)NBT * CE), 256, 0, stream>>>(x, TMP, out_by, NBT * CE);

    // ---- fusion ----
    k_avg<<<egrid((long long)NBT * CE), 256, 0, stream>>>(out_fy, out_by, bufA, NBT * CE);
    mlp2(bufA, NBT, IN(85), IN(86), IN(87), IN(88), IN(89), IN(90),
         nullptr, nullptr, bufB, bufC, out_fb);

    // ---- clip attention pooling ----
    k_rowdot<<<dim3((unsigned)(((long long)NBT * 32 + 255) / 256)), 256, 0, stream>>>(
        out_fb, IN(91), IN(92), WTA, NBT, CE);
    mlp2(out_fb, NBT, IN(93), IN(94), IN(95), IN(96), IN(97), IN(98),
         nullptr, nullptr, bufA, bufB, FEAT);
    k_clipcomb<<<egrid((long long)(NBT / 5) * CE), 256, 0, stream>>>(
        FEAT, WTA, out_feat, (NBT / 5) * CE, CE);
}
